// LSTMClassifier_23398981829079
// MI455X (gfx1250) — matrix operand order repaired and checked by reading, not compiled, not run on hardware
//
#include <hip/hip_runtime.h>

typedef __attribute__((ext_vector_type(16))) _Float16 v16h;
typedef __attribute__((ext_vector_type(8)))  _Float16 v8h;
typedef __attribute__((ext_vector_type(8)))  float    v8f;

#define VOCAB   50000
#define E_DIM   300
#define E_PAD   320     // padded to multiple of 32 (K of wmma f16)
#define H_DIM   1024
#define G4      4096    // 4*H
#define OUT_DIM 5
#define B_SZ    64
#define T_SZ    256
#define PAD_IDX 1

// ---------------------------------------------------------------------------
// WMMA fragment loaders (wave32; contiguous k order, same in A and B)
// ---------------------------------------------------------------------------

// A-matrix 16x32 f16 (MxK). Lane l holds row M = m0 + (l&15).
// halves 0..15 : K = k + (l<16 ? 0 : 16) .. +15
__device__ inline v16h load_a_frag(const _Float16* __restrict__ base,
                                   int rowStride, int m0, int k, int lane) {
    int M    = m0 + (lane & 15);
    int koff = (lane & 16) ? 16 : 0;
    const _Float16* p = base + (size_t)M * rowStride + k + koff;
    return *(const v16h*)(p);         // K = k+koff .. +15
}

// B-matrix 32x16 f16 (KxN) where B[k][n] = W[n][k], W row-major [N][K].
// Lane l holds column n = n0 + (l&15); lanes 0-15: K=k..k+15, lanes 16-31: K=k+16..k+31.
// One contiguous 32-byte load per lane.
__device__ inline v16h load_b_frag(const _Float16* __restrict__ w,
                                   int Kstride, int n0, int k, int lane) {
    int n    = n0 + (lane & 15);
    int koff = (lane & 16) ? 16 : 0;
    return *(const v16h*)(w + (size_t)n * Kstride + k + koff);
}

__device__ inline float sigmoidf_(float x) { return 1.0f / (1.0f + __expf(-x)); }

// ---------------------------------------------------------------------------
// Prep kernels
// ---------------------------------------------------------------------------

// W_ih [4H, E] f32 -> f16 padded to [4H, E_PAD]
__global__ void convert_wih_kernel(const float* __restrict__ W, _Float16* __restrict__ W16) {
    int i = blockIdx.x * blockDim.x + threadIdx.x;
    if (i >= G4 * E_PAD) return;
    int row = i / E_PAD, e = i % E_PAD;
    W16[i] = (e < E_DIM) ? (_Float16)W[(size_t)row * E_DIM + e] : (_Float16)0.0f;
}

// W_hh [4H, H] f32 -> f16
__global__ void convert_whh_kernel(const float* __restrict__ W, _Float16* __restrict__ W16) {
    int i = blockIdx.x * blockDim.x + threadIdx.x;
    if (i >= G4 * H_DIM) return;
    W16[i] = (_Float16)W[i];
}

// gbias = b_ih + b_hh; zero h ping-pong buffers, c, final hidden; compute lengths
__global__ void prep_state_kernel(const float* __restrict__ bih, const float* __restrict__ bhh,
                                  const int* __restrict__ x,
                                  float* __restrict__ gbias, int* __restrict__ lens,
                                  _Float16* __restrict__ h16a, _Float16* __restrict__ h16b,
                                  float* __restrict__ cbuf, float* __restrict__ finalh) {
    int i = blockIdx.x * blockDim.x + threadIdx.x;
    if (i < B_SZ * H_DIM) {
        h16a[i]   = (_Float16)0.0f;
        h16b[i]   = (_Float16)0.0f;
        cbuf[i]   = 0.0f;
        finalh[i] = 0.0f;
    }
    if (i < G4) gbias[i] = bih[i] + bhh[i];
    if (i < B_SZ) {
        int c = 0;
        for (int t = 0; t < T_SZ; ++t) c += (x[i * T_SZ + t] != PAD_IDX);
        lens[i] = c;
    }
}

// embedding lookup -> f16, padded to E_PAD columns
__global__ void embed_kernel(const int* __restrict__ x, const float* __restrict__ emb,
                             _Float16* __restrict__ inp16) {
    const size_t N = (size_t)B_SZ * T_SZ * E_PAD;
    for (size_t i = (size_t)blockIdx.x * blockDim.x + threadIdx.x; i < N;
         i += (size_t)gridDim.x * blockDim.x) {
        int    e   = (int)(i % E_PAD);
        size_t row = i / E_PAD;   // = b*T + t
        float  v   = 0.0f;
        if (e < E_DIM) v = emb[(size_t)x[row] * E_DIM + e];
        inp16[i] = (_Float16)v;
    }
}

// ---------------------------------------------------------------------------
// Fused LSTM step: gates = h@Whh^T + inp_t@Wih^T + (b_ih+b_hh); cell update.
// 256 waves: mtile (4) x jtile (64). Each wave: 16x16 hidden tile, all 4 gates.
// ---------------------------------------------------------------------------
__global__ __launch_bounds__(256) void lstm_step_kernel(
    const _Float16* __restrict__ inp16, const _Float16* __restrict__ Wih16,
    const _Float16* __restrict__ Whh16, const float* __restrict__ gbias,
    const int* __restrict__ x, const int* __restrict__ lens,
    const _Float16* __restrict__ h_in, _Float16* __restrict__ h_out,
    float* __restrict__ cbuf, float* __restrict__ finalh, int t)
{
    int lane  = threadIdx.x & 31;
    int w     = blockIdx.x * (blockDim.x >> 5) + (threadIdx.x >> 5);
    int mtile = w >> 6;                 // 0..3   (batch tile)
    int jtile = w & 63;                 // 0..63  (hidden tile)
    int m0    = mtile * 16;
    int j0    = jtile * 16;

    v8f acc[4] = { v8f{}, v8f{}, v8f{}, v8f{} };   // i, f, g, o accumulators

    // Recurrent contribution: K = H_DIM
    for (int k = 0; k < H_DIM; k += 32) {
        v16h a = load_a_frag(h_in, H_DIM, m0, k, lane);
#pragma unroll
        for (int g = 0; g < 4; ++g) {
            v16h b = load_b_frag(Whh16, H_DIM, g * H_DIM + j0, k, lane);
            acc[g] = __builtin_amdgcn_wmma_f32_16x16x32_f16(
                false, a, false, b, (short)0, acc[g], false, false);
        }
    }

    // Input projection contribution: K = E_PAD, A rows are inp16[(b*T + t), :]
    const _Float16* abase = inp16 + (size_t)t * E_PAD;   // + b*(T*E_PAD) via rowStride
    for (int k = 0; k < E_PAD; k += 32) {
        v16h a = load_a_frag(abase, T_SZ * E_PAD, m0, k, lane);
#pragma unroll
        for (int g = 0; g < 4; ++g) {
            v16h b = load_b_frag(Wih16, E_PAD, g * H_DIM + j0, k, lane);
            acc[g] = __builtin_amdgcn_wmma_f32_16x16x32_f16(
                false, a, false, b, (short)0, acc[g], false, false);
        }
    }

    // Epilogue: bias, activations, cell/hidden update (C/D layout: M=r+8*lane[4], N=lane[3:0])
    int   col = j0 + (lane & 15);
    float bi  = gbias[0 * H_DIM + col];
    float bf  = gbias[1 * H_DIM + col];
    float bg  = gbias[2 * H_DIM + col];
    float bo  = gbias[3 * H_DIM + col];
    int rbase = m0 + ((lane & 16) ? 8 : 0);

#pragma unroll
    for (int r = 0; r < 8; ++r) {
        int    Mr  = rbase + r;
        float  iv  = sigmoidf_(acc[0][r] + bi);
        float  fv  = sigmoidf_(acc[1][r] + bf);
        float  gv  = tanhf(acc[2][r] + bg);
        float  ov  = sigmoidf_(acc[3][r] + bo);
        size_t off = (size_t)Mr * H_DIM + col;
        float  cn  = fv * cbuf[off] + iv * gv;
        cbuf[off]  = cn;
        float  hv  = ov * tanhf(cn);
        h_out[off] = (_Float16)hv;
        if (t == lens[Mr] - 1) {
            bool pad    = (x[Mr * T_SZ + t] == PAD_IDX);
            finalh[off] = pad ? 0.0f : hv;
        }
    }
}

// ---------------------------------------------------------------------------
// logits = finalh @ W_out^T + b_out   (64x1024 x 1024x5, tiny -> plain f32)
// ---------------------------------------------------------------------------
__global__ __launch_bounds__(32) void logits_kernel(const float* __restrict__ finalh,
                                                    const float* __restrict__ Wout,
                                                    const float* __restrict__ bout,
                                                    float* __restrict__ out) {
    int b    = blockIdx.x;
    int lane = threadIdx.x;
    for (int o = 0; o < OUT_DIM; ++o) {
        float s = 0.0f;
        for (int k = lane; k < H_DIM; k += 32)
            s += finalh[(size_t)b * H_DIM + k] * Wout[(size_t)o * H_DIM + k];
        for (int m = 16; m > 0; m >>= 1) s += __shfl_xor(s, m, 32);
        if (lane == 0) out[b * OUT_DIM + o] = s + bout[o];
    }
}

// ---------------------------------------------------------------------------

extern "C" void kernel_launch(void* const* d_in, const int* in_sizes, int n_in,
                              void* d_out, int out_size, void* d_ws, size_t ws_size,
                              hipStream_t stream) {
    const int*   x    = (const int*)d_in[0];
    const float* emb  = (const float*)d_in[1];
    const float* Wih  = (const float*)d_in[2];
    const float* bih  = (const float*)d_in[3];
    const float* Whh  = (const float*)d_in[4];
    const float* bhh  = (const float*)d_in[5];
    const float* Wout = (const float*)d_in[6];
    const float* bout = (const float*)d_in[7];
    float* out = (float*)d_out;

    // Workspace carve (256B aligned): ~22 MB total
    char* ws = (char*)d_ws;
    auto carve = [&](size_t bytes) {
        char* p = ws;
        ws += (bytes + 255) & ~(size_t)255;
        return p;
    };
    _Float16* Wih16  = (_Float16*)carve((size_t)G4 * E_PAD * sizeof(_Float16));
    _Float16* Whh16  = (_Float16*)carve((size_t)G4 * H_DIM * sizeof(_Float16));
    _Float16* inp16  = (_Float16*)carve((size_t)B_SZ * T_SZ * E_PAD * sizeof(_Float16));
    _Float16* h16a   = (_Float16*)carve((size_t)B_SZ * H_DIM * sizeof(_Float16));
    _Float16* h16b   = (_Float16*)carve((size_t)B_SZ * H_DIM * sizeof(_Float16));
    float*    cbuf   = (float*)carve((size_t)B_SZ * H_DIM * sizeof(float));
    float*    finalh = (float*)carve((size_t)B_SZ * H_DIM * sizeof(float));
    float*    gbias  = (float*)carve((size_t)G4 * sizeof(float));
    int*      lens   = (int*)carve((size_t)B_SZ * sizeof(int));

    // Weight conversion (one-time per call; tiny vs. the 256-step loop)
    convert_wih_kernel<<<(G4 * E_PAD + 255) / 256, 256, 0, stream>>>(Wih, Wih16);
    convert_whh_kernel<<<(G4 * H_DIM + 255) / 256, 256, 0, stream>>>(Whh, Whh16);
    prep_state_kernel<<<(B_SZ * H_DIM + 255) / 256, 256, 0, stream>>>(
        bih, bhh, x, gbias, lens, h16a, h16b, cbuf, finalh);
    embed_kernel<<<2048, 256, 0, stream>>>(x, emb, inp16);

    // Sequential recurrence: one fused GEMM+cell kernel per step; ping-pong h.
    _Float16* hin  = h16a;
    _Float16* hout = h16b;
    for (int t = 0; t < T_SZ; ++t) {
        lstm_step_kernel<<<32, 256, 0, stream>>>(inp16, Wih16, Whh16, gbias, x, lens,
                                                 hin, hout, cbuf, finalh, t);
        _Float16* tmp = hin; hin = hout; hout = tmp;
    }

    logits_kernel<<<B_SZ, 32, 0, stream>>>(finalh, Wout, bout, out);
}
